// GraphAttentionLayer_13005160972785
// MI455X (gfx1250) — matrix operand order repaired and checked by reading, not compiled, not run on hardware
//
#include <hip/hip_runtime.h>

typedef _Float16 h16;
typedef __attribute__((ext_vector_type(16))) _Float16 v16h;
typedef __attribute__((ext_vector_type(8)))  _Float16 v8h;
typedef __attribute__((ext_vector_type(4)))  _Float16 v4h;
typedef __attribute__((ext_vector_type(8)))  float    v8f;
typedef __attribute__((ext_vector_type(4)))  float    v4f;
typedef __attribute__((ext_vector_type(4)))  int      v4i;

static __device__ __forceinline__ v16h cat8(v8h lo, v8h hi) {
    return __builtin_shufflevector(lo, hi, 0,1,2,3,4,5,6,7,8,9,10,11,12,13,14,15);
}

// -----------------------------------------------------------------------------
// Kernel 1: h = x @ W (f16 WMMA, f32 accum), fused si = h.a1, sj = h.a2.
// Block: 32 rows x 256 cols, 8 waves; wave w -> rowgrp = w>>2, head(colgrp) = w&3.
// -----------------------------------------------------------------------------
__global__ __launch_bounds__(256) void gat_xw_kernel(
    const float* __restrict__ x, const float* __restrict__ W,
    const float* __restrict__ a, h16* __restrict__ hf,
    float* __restrict__ si, float* __restrict__ sj)
{
    __shared__ h16 xl[32 * 40];     // x tile, row-major [row][k], stride 40 halves (80B)
    __shared__ h16 wl[256 * 40];    // W tile, transposed [col][k], stride 40 halves

    const int t    = threadIdx.x;
    const int lane = t & 31;
    const int w    = t >> 5;
    const int n0   = blockIdx.x * 32;
    const int rowgrp = w >> 2;          // 0..1  (16 rows each)
    const int head   = w & 3;           // 0..3  (64 cols each)
    const int hb8    = (lane >> 4) * 8; // row offset of upper half-lanes in C
    const int arow   = rowgrp * 16 + (lane & 15);
    const int k0     = (lane >> 4) * 16;

    v8f acc[4] = {};

    for (int kc = 0; kc < 8; ++kc) {
        const int kb = kc * 32;
        __syncthreads();
        // ---- stage x tile (f32 -> f16), row-major
        {
            int r = t >> 3, c4 = (t & 7) * 4;
            v4f xv = *(const v4f*)(x + (n0 + r) * 256 + kb + c4);
            v4h pk = { (h16)xv[0], (h16)xv[1], (h16)xv[2], (h16)xv[3] };
            *(v4h*)&xl[r * 40 + c4] = pk;
        }
        // ---- stage W tile transposed [col][k]: 4 k-rows x 8 cols per thread,
        //      packed into b64 LDS stores along the k (fast) axis.
        {
            int r4 = (t >> 5) * 4;      // k rows r4..r4+3
            int c8 = (t & 31) * 8;      // cols  c8..c8+7
            h16 hv[4][8];
            #pragma unroll
            for (int q = 0; q < 4; ++q) {
                v4f w0 = *(const v4f*)(W + (kb + r4 + q) * 256 + c8);
                v4f w1 = *(const v4f*)(W + (kb + r4 + q) * 256 + c8 + 4);
                #pragma unroll
                for (int e = 0; e < 4; ++e) {
                    hv[q][e]     = (h16)w0[e];
                    hv[q][4 + e] = (h16)w1[e];
                }
            }
            #pragma unroll
            for (int e = 0; e < 8; ++e) {
                v4h pk = { hv[0][e], hv[1][e], hv[2][e], hv[3][e] };
                *(v4h*)&wl[(c8 + e) * 40 + r4] = pk;
            }
        }
        __syncthreads();
        // ---- A operand: row = lane&15 (+rowgrp), K runs {hb8..hb8+7, 16+hb8..}
        v8h alo = *(const v8h*)&xl[arow * 40 + hb8];
        v8h ahi = *(const v8h*)&xl[arow * 40 + 16 + hb8];
        v16h A = cat8(alo, ahi);
        #pragma unroll
        for (int tt = 0; tt < 4; ++tt) {
            int col = head * 64 + tt * 16 + (lane & 15);
            v8h blo = *(const v8h*)&wl[col * 40 + hb8];
            v8h bhi = *(const v8h*)&wl[col * 40 + 16 + hb8];
            v16h B = cat8(blo, bhi);
            acc[tt] = __builtin_amdgcn_wmma_f32_16x16x32_f16(
                false, A, false, B, (short)0, acc[tt], false, false);
        }
    }

    // ---- write h (f16) + fused si/sj reductions
    float p1[8] = {}, p2[8] = {};
    #pragma unroll
    for (int tt = 0; tt < 4; ++tt) {
        int dloc = tt * 16 + (lane & 15);    // d within head: 0..63
        float a1v = a[dloc];
        float a2v = a[64 + dloc];
        #pragma unroll
        for (int r = 0; r < 8; ++r) {
            float v = acc[tt][r];
            int row = n0 + rowgrp * 16 + r + hb8;
            hf[row * 256 + head * 64 + dloc] = (h16)v;
            p1[r] += v * a1v;
            p2[r] += v * a2v;
        }
    }
    #pragma unroll
    for (int r = 0; r < 8; ++r) {
        float s1 = p1[r], s2 = p2[r];
        #pragma unroll
        for (int m = 1; m < 16; m <<= 1) {
            s1 += __shfl_xor(s1, m, 32);
            s2 += __shfl_xor(s2, m, 32);
        }
        int row = n0 + rowgrp * 16 + r + hb8;
        if ((lane & 15) == 0) {
            si[row * 4 + head] = s1;
            sj[row * 4 + head] = s2;
        }
    }
}

// -----------------------------------------------------------------------------
// Kernel 2: flash-attention over adj mask + fused LayerNorm.
// Block: 32 rows (n) x all 4 heads; wave w -> rowgrp = w>>2, head = w&3.
// Each wave owns a full head (16 rows x 64 cols = 4 accum tiles): one P/softmax
// computation feeds 4 WMMAs. Streams m in chunks of 32.
// -----------------------------------------------------------------------------
__global__ __launch_bounds__(256) void gat_attn_kernel(
    const int* __restrict__ adj, const h16* __restrict__ hf,
    const float* __restrict__ si, const float* __restrict__ sj,
    const float* __restrict__ gamma, const float* __restrict__ beta,
    float* __restrict__ out)
{
    __shared__ h16   ht[256 * 40];    // h tile transposed [col][k], stride 40 halves
    __shared__ int   adjl[32 * 32];   // adj tile [row][m]
    __shared__ float sjl[4 * 32];     // sj tile  [head][m]
    __shared__ float outl[32 * 256];  // output staging for LayerNorm

    const int t    = threadIdx.x;
    const int lane = t & 31;
    const int w    = t >> 5;
    const int n0   = blockIdx.x * 32;
    const int rowgrp = w >> 2;           // 0..1
    const int head   = w & 3;            // 0..3
    const int row    = rowgrp * 16 + (lane & 15);  // block-local n row for P
    const int hb8    = (lane >> 4) * 8;
    const int k0     = (lane >> 4) * 16;

    const float si_v = si[(n0 + row) * 4 + head];
    float m_run = -1e30f, l_run = 0.f;
    v8f acc[4] = {};

    for (int mc = 0; mc < 128; ++mc) {
        const int m0 = mc * 32;
        __syncthreads();
        // ---- stage adj tile (32x32 int), vectorized
        {
            int r = t >> 3, c = (t & 7) * 4;
            *(v4i*)&adjl[r * 32 + c] = *(const v4i*)(adj + (n0 + r) * 4096 + m0 + c);
        }
        // ---- stage sj tile [head][m]
        if (t < 128) {
            int k = t >> 2, hh = t & 3;
            sjl[hh * 32 + k] = sj[(m0 + k) * 4 + hh];
        }
        // ---- stage h tile transposed [col][k]: 4 m-rows x 8 cols per thread,
        //      b64-packed stores along the k (fast) axis.
        {
            int r4 = (t >> 5) * 4;      // rows r4..r4+3
            int c8 = (t & 31) * 8;      // cols c8..c8+7
            v8h rv0 = *(const v8h*)(hf + (m0 + r4 + 0) * 256 + c8);
            v8h rv1 = *(const v8h*)(hf + (m0 + r4 + 1) * 256 + c8);
            v8h rv2 = *(const v8h*)(hf + (m0 + r4 + 2) * 256 + c8);
            v8h rv3 = *(const v8h*)(hf + (m0 + r4 + 3) * 256 + c8);
            #pragma unroll
            for (int e = 0; e < 8; ++e) {
                v4h pk = { rv0[e], rv1[e], rv2[e], rv3[e] };
                *(v4h*)&ht[(c8 + e) * 40 + r4] = pk;
            }
            if (mc + 1 < 128)
                __builtin_prefetch(hf + (m0 + 32 + r4) * 256 + c8, 0, 1);
        }
        __syncthreads();

        // ---- score tile in WMMA-A layout: K(lane,j) = (j>>3)*16 + hb8 + (j&7)
        float p[16];
        float cm = -1e30f;
        #pragma unroll
        for (int run = 0; run < 2; ++run) {
            int kbb = run * 16 + hb8;
            v4i a0 = *(const v4i*)&adjl[row * 32 + kbb];
            v4i a1 = *(const v4i*)&adjl[row * 32 + kbb + 4];
            v4f s0 = *(const v4f*)&sjl[head * 32 + kbb];
            v4f s1 = *(const v4f*)&sjl[head * 32 + kbb + 4];
            #pragma unroll
            for (int i = 0; i < 8; ++i) {
                int   av = (i < 4) ? a0[i] : a1[i - 4];
                float sv = (i < 4) ? s0[i] : s1[i - 4];
                float e = si_v + sv;
                e = (e > 0.f) ? e : 0.2f * e;          // leaky relu
                e = (av > 0) ? e : -1e30f;             // adjacency mask
                p[run * 8 + i] = e;
                cm = fmaxf(cm, e);
            }
        }
        cm = fmaxf(cm, __shfl_xor(cm, 16, 32));        // combine K-halves per row
        float m_new = fmaxf(m_run, cm);

        // ---- rescale accumulators only if some row got a new max (wave-uniform)
        float scale = 1.f;
        if (__ballot(m_new > m_run) != 0ull) {
            scale = __expf(m_run - m_new);
            #pragma unroll
            for (int r = 0; r < 8; ++r) {
                float sc = __shfl(scale, r + hb8, 32);
                #pragma unroll
                for (int tt = 0; tt < 4; ++tt) acc[tt][r] *= sc;
            }
        }
        m_run = m_new;

        float rsum = 0.f;
        #pragma unroll
        for (int j = 0; j < 16; ++j) {
            float pv = (p[j] > -1e29f) ? __expf(p[j] - m_new) : 0.f;
            p[j] = pv;
            rsum += pv;
        }
        rsum += __shfl_xor(rsum, 16, 32);
        l_run = l_run * scale + rsum;

        // ---- A operand = exp weights (f16)
        v16h A;
        #pragma unroll
        for (int j = 0; j < 16; ++j) A[j] = (h16)p[j];
        // ---- 4 WMMAs: one P tile feeds the whole head (16 n x 64 d x 32 m)
        #pragma unroll
        for (int tt = 0; tt < 4; ++tt) {
            int col = head * 64 + tt * 16 + (lane & 15);
            v8h blo = *(const v8h*)&ht[col * 40 + hb8];
            v8h bhi = *(const v8h*)&ht[col * 40 + 16 + hb8];
            acc[tt] = __builtin_amdgcn_wmma_f32_16x16x32_f16(
                false, A, false, cat8(blo, bhi), (short)0, acc[tt], false, false);
        }
    }

    // ---- normalize by softmax denominator, stage to LDS for LayerNorm
    float inv = (l_run > 0.f) ? (1.f / l_run) : 0.f;
    #pragma unroll
    for (int r = 0; r < 8; ++r) {
        float iv = __shfl(inv, r + hb8, 32);
        int orow = rowgrp * 16 + r + hb8;
        #pragma unroll
        for (int tt = 0; tt < 4; ++tt) {
            int c = head * 64 + tt * 16 + (lane & 15);
            outl[orow * 256 + c] = acc[tt][r] * iv;
        }
    }
    __syncthreads();

    // ---- fused LayerNorm over F=256 (8 threads per row, contiguous 32-col segs)
    {
        int r = t >> 3, seg = t & 7;
        float sum = 0.f, ss = 0.f;
        #pragma unroll
        for (int q = 0; q < 8; ++q) {
            v4f v = *(const v4f*)&outl[r * 256 + seg * 32 + q * 4];
            #pragma unroll
            for (int e = 0; e < 4; ++e) { sum += v[e]; ss += v[e] * v[e]; }
        }
        #pragma unroll
        for (int m = 1; m < 8; m <<= 1) {
            sum += __shfl_xor(sum, m, 32);
            ss  += __shfl_xor(ss,  m, 32);
        }
        float mean = sum * (1.f / 256.f);
        float var  = ss * (1.f / 256.f) - mean * mean;
        float rinv = rsqrtf(var + 1e-5f);
        #pragma unroll
        for (int q = 0; q < 8; ++q) {
            int c = seg * 32 + q * 4;
            v4f v = *(const v4f*)&outl[r * 256 + c];
            v4f g = *(const v4f*)(gamma + c);
            v4f b = *(const v4f*)(beta + c);
            v4f res;
            #pragma unroll
            for (int e = 0; e < 4; ++e) res[e] = g[e] * (v[e] - mean) * rinv + b[e];
            *(v4f*)(out + (n0 + r) * 256 + c) = res;
        }
    }
}

extern "C" void kernel_launch(void* const* d_in, const int* in_sizes, int n_in,
                              void* d_out, int out_size, void* d_ws, size_t ws_size,
                              hipStream_t stream) {
    (void)in_sizes; (void)n_in; (void)out_size; (void)ws_size;
    const float* x     = (const float*)d_in[0];
    const int*   adj   = (const int*)d_in[1];
    const float* W     = (const float*)d_in[2];
    const float* a     = (const float*)d_in[3];
    const float* gamma = (const float*)d_in[4];
    const float* beta  = (const float*)d_in[5];
    float* out = (float*)d_out;

    char* ws = (char*)d_ws;
    h16*   hf = (h16*)ws;                           // 4096*256 f16 = 2 MB
    float* si = (float*)(ws + 4096 * 256 * 2);      // 4096*4 f32
    float* sj = si + 4096 * 4;                      // 4096*4 f32

    gat_xw_kernel<<<128, 256, 0, stream>>>(x, W, a, hf, si, sj);
    gat_attn_kernel<<<128, 256, 0, stream>>>(adj, hf, si, sj, gamma, beta, out);
}
